// Model_7610682048728
// MI455X (gfx1250) — matrix operand order repaired and checked by reading, not compiled, not run on hardware
//
#include <hip/hip_runtime.h>
#include <hip/hip_bf16.h>
#include <math.h>

typedef _Float16 v16h __attribute__((ext_vector_type(16)));
typedef float    v8f  __attribute__((ext_vector_type(8)));

#define B_     32
#define L_     512
#define E_     512
#define NMARK  4
#define D_     512
#define H_     8
#define DFF_   2048
#define NLAY   2
#define BKT    4
#define NHASH  4
#define NTOK   516
#define TPAD   520
#define DH     64
#define NB     130          // buckets per hash round
#define NCH    520          // chunks = NHASH * TPAD / BKT
#define BH     (B_*H_)      // 256
#define TT     (NHASH*TPAD) // 2080
#define SORTN  4096
#define ROWS   (B_*TPAD)    // 16640

// ---------------------------------------------------------------- RevIN stats
__global__ __launch_bounds__(256)
void k_stats(const float* __restrict__ x_enc, float* __restrict__ meanb,
             float* __restrict__ stdb) {
    int b = blockIdx.x;
    int e = blockIdx.y * 256 + threadIdx.x;
    const float* p = x_enc + (size_t)b * L_ * E_ + e;
    float s = 0.f, ss = 0.f;
    for (int l = 0; l < L_; ++l) {
        float v = p[(size_t)l * E_];
        s += v; ss += v * v;
    }
    float m  = s * (1.f / L_);
    float va = ss * (1.f / L_) - m * m;
    meanb[b * E_ + e] = m;
    stdb [b * E_ + e] = sqrtf(va + 1e-5f);
}

// ------------------------------------------------------- build inverted tokens
__global__ __launch_bounds__(256)
void k_build_tok(const float* __restrict__ x_enc, const float* __restrict__ x_mark,
                 const float* __restrict__ meanb, const float* __restrict__ stdb,
                 float* __restrict__ tok) {
    size_t i = (size_t)blockIdx.x * 256 + threadIdx.x;  // over B*520*512
    int b   = (int)(i / ((size_t)TPAD * L_));
    int rem = (int)(i % ((size_t)TPAD * L_));
    int n = rem / L_;
    int l = rem % L_;
    float v;
    if (n < E_) {
        v = (x_enc[(size_t)b * L_ * E_ + (size_t)l * E_ + n] - meanb[b * E_ + n])
            / stdb[b * E_ + n];
    } else if (n < NTOK) {
        v = x_mark[(size_t)b * L_ * NMARK + (size_t)l * NMARK + (n - E_)];
    } else {
        v = 0.f;
    }
    tok[i] = v;
}

// ----------------------------------------------------- WMMA GEMM: C = A @ W^T
// A: [M,K] f32 row-major, W: [N,K] f32 row-major, C: [M,N] f32.
// 256 threads = 8 waves; macro-tile 128(M) x 64(N); each wave owns a 32x32 C
// tile = 2x2 fragments -> 4 v_wmma_f32_16x16x32_f16 per K-step with A/B reuse.
// f32 -> f16 staged in LDS; f32 accumulation.
__global__ __launch_bounds__(256)
void k_gemm(const float* __restrict__ A, const float* __restrict__ W,
            const float* __restrict__ bias, float* __restrict__ C,
            int M, int N, int K, int relu) {
    __shared__ _Float16 As[128][40];
    __shared__ _Float16 Ws[64][40];
    const int tid  = threadIdx.x;
    const int lane = tid & 31;
    const int w    = tid >> 5;
    const int wm   = w & 3, wn = w >> 2;   // wave tile: rows wm*32, cols wn*32
    const int n0   = blockIdx.x * 64;
    const int m0   = blockIdx.y * 128;

    // A-tile loader mapping: 128 rows x 32 cols, 2 threads/row, 16 floats each
    const int larow = tid >> 1;            // 0..127
    const int lacol = (tid & 1) * 16;      // 0 or 16
    // W-tile loader mapping: 64 rows x 32 cols, 4 threads/row, 8 floats each
    const int lwrow = tid >> 2;            // 0..63
    const int lwcol = (tid & 3) * 8;       // 0,8,16,24

    v8f c00 = {}; v8f c01 = {}; v8f c10 = {}; v8f c11 = {};

    const int mr0 = wm * 32 + (lane & 15); // A frag rows (lo 16)
    const int mr1 = mr0 + 16;              // A frag rows (hi 16)
    const int kA  = (lane >> 4) * 8;       // A frag K sub-base
    const int kB  = (lane >> 4) * 16;      // unused; B frags also index with kk from kA
    const int nco = lane & 15;

    for (int k0 = 0; k0 < K; k0 += 32) {
        const float* ap = A + (size_t)(m0 + larow) * K + k0 + lacol;
        const float* wp = W + (size_t)(n0 + lwrow) * K + k0 + lwcol;
        float4 a0 = *(const float4*)ap;
        float4 a1 = *(const float4*)(ap + 4);
        float4 a2 = *(const float4*)(ap + 8);
        float4 a3 = *(const float4*)(ap + 12);
        float4 w0 = *(const float4*)wp;
        float4 w1 = *(const float4*)(wp + 4);
        if (k0 + 32 < K) {
            __builtin_prefetch(ap + 32, 0, 1);   // global_prefetch_b8
            __builtin_prefetch(wp + 32, 0, 1);
        }
        __syncthreads();
        As[larow][lacol + 0]  = (_Float16)a0.x; As[larow][lacol + 1]  = (_Float16)a0.y;
        As[larow][lacol + 2]  = (_Float16)a0.z; As[larow][lacol + 3]  = (_Float16)a0.w;
        As[larow][lacol + 4]  = (_Float16)a1.x; As[larow][lacol + 5]  = (_Float16)a1.y;
        As[larow][lacol + 6]  = (_Float16)a1.z; As[larow][lacol + 7]  = (_Float16)a1.w;
        As[larow][lacol + 8]  = (_Float16)a2.x; As[larow][lacol + 9]  = (_Float16)a2.y;
        As[larow][lacol + 10] = (_Float16)a2.z; As[larow][lacol + 11] = (_Float16)a2.w;
        As[larow][lacol + 12] = (_Float16)a3.x; As[larow][lacol + 13] = (_Float16)a3.y;
        As[larow][lacol + 14] = (_Float16)a3.z; As[larow][lacol + 15] = (_Float16)a3.w;
        Ws[lwrow][lwcol + 0] = (_Float16)w0.x; Ws[lwrow][lwcol + 1] = (_Float16)w0.y;
        Ws[lwrow][lwcol + 2] = (_Float16)w0.z; Ws[lwrow][lwcol + 3] = (_Float16)w0.w;
        Ws[lwrow][lwcol + 4] = (_Float16)w1.x; Ws[lwrow][lwcol + 5] = (_Float16)w1.y;
        Ws[lwrow][lwcol + 6] = (_Float16)w1.z; Ws[lwrow][lwcol + 7] = (_Float16)w1.w;
        __syncthreads();

        v16h af0, af1, bf0, bf1;
#pragma unroll
        for (int h = 0; h < 16; ++h) {
            int kk = ((h >> 3) << 4) + kA + (h & 7);  // 16*(h/8) + 8*(lane/16) + h%8
            af0[h] = As[mr0][kk];
            af1[h] = As[mr1][kk];
            bf0[h] = Ws[wn * 32 + nco][kk];
            bf1[h] = Ws[wn * 32 + 16 + nco][kk];
        }
        c00 = __builtin_amdgcn_wmma_f32_16x16x32_f16(false, af0, false, bf0,
                                                     (short)0, c00, false, false);
        c01 = __builtin_amdgcn_wmma_f32_16x16x32_f16(false, af0, false, bf1,
                                                     (short)0, c01, false, false);
        c10 = __builtin_amdgcn_wmma_f32_16x16x32_f16(false, af1, false, bf0,
                                                     (short)0, c10, false, false);
        c11 = __builtin_amdgcn_wmma_f32_16x16x32_f16(false, af1, false, bf1,
                                                     (short)0, c11, false, false);
    }
    __syncthreads();

    int col0 = n0 + wn * 32 + nco;
    int col1 = col0 + 16;
    float b0v = bias ? bias[col0] : 0.f;
    float b1v = bias ? bias[col1] : 0.f;
#pragma unroll
    for (int r = 0; r < 8; ++r) {
        int row0 = m0 + wm * 32 + r + ((lane >> 4) << 3);
        int row1 = row0 + 16;
        float v00 = c00[r] + b0v;
        float v01 = c01[r] + b1v;
        float v10 = c10[r] + b0v;
        float v11 = c11[r] + b1v;
        if (relu) {
            v00 = fmaxf(v00, 0.f); v01 = fmaxf(v01, 0.f);
            v10 = fmaxf(v10, 0.f); v11 = fmaxf(v11, 0.f);
        }
        C[(size_t)row0 * N + col0] = v00;
        C[(size_t)row0 * N + col1] = v01;
        C[(size_t)row1 * N + col0] = v10;
        C[(size_t)row1 * N + col1] = v11;
    }
}

// ------------------------------------------------------ zero padded token rows
__global__ __launch_bounds__(256)
void k_zero_pad(float* __restrict__ x) {
    int i   = blockIdx.x * 256 + threadIdx.x;    // B*4*512
    int b   = i / (4 * D_);
    int rem = i % (4 * D_);
    x[((size_t)(b * TPAD + NTOK + rem / D_)) * D_ + (rem % D_)] = 0.f;
}

// --------------------------------------------- LSH bucket keys (rot + argmax)
__global__ __launch_bounds__(256)
void k_buckets(const float* __restrict__ qkbuf, const float* __restrict__ rot,
               unsigned long long* __restrict__ sortbuf) {
    int bh = blockIdx.x;
    int j  = blockIdx.y * 256 + threadIdx.x;     // 0..4095
    unsigned long long* out = sortbuf + (size_t)bh * SORTN;
    if (j >= TT) { out[j] = ~0ull; return; }
    int h = j / TPAD;
    int t = j - h * TPAD;
    int b = bh >> 3, head = bh & 7;
    const float* q = qkbuf + ((size_t)(b * TPAD + t)) * D_ + head * DH;
    float qr[DH];
#pragma unroll
    for (int f = 0; f < DH; ++f) qr[f] = q[f];
    const float* rp = rot + h * (NB / 2);        // rot[f][h][i] : f*260 + h*65 + i
    float best = -3.0e38f; int bi = 0;
    for (int i = 0; i < NB / 2; ++i) {
        float acc = 0.f;
#pragma unroll 8
        for (int f = 0; f < DH; ++f) acc += qr[f] * rp[f * (NHASH * (NB / 2)) + i];
        if ( acc > best) { best =  acc; bi = i; }
        if (-acc > best) { best = -acc; bi = i + NB / 2; }
    }
    unsigned key = (unsigned)((bi + h * NB) * TPAD + t);
    out[j] = ((unsigned long long)key << 32) | (unsigned)j;
}

// ------------------------------------------------ per-(b,h) bitonic sort (LDS)
__global__ __launch_bounds__(256)
void k_sort(const unsigned long long* __restrict__ sortbuf,
            unsigned* __restrict__ stbuf, unsigned* __restrict__ undo) {
    __shared__ unsigned long long s[SORTN];
    int bh = blockIdx.x, tid = threadIdx.x;
    const unsigned long long* in = sortbuf + (size_t)bh * SORTN;
    for (int i = tid; i < SORTN; i += 256) s[i] = in[i];
    __syncthreads();
    for (int k = 2; k <= SORTN; k <<= 1) {
        for (int jj = k >> 1; jj > 0; jj >>= 1) {
            for (int t = tid; t < SORTN / 2; t += 256) {
                int i   = ((t / jj) * (jj << 1)) + (t % jj);
                int ixj = i + jj;
                bool up = ((i & k) == 0);
                unsigned long long a = s[i], bb = s[ixj];
                if ((a > bb) == up) { s[i] = bb; s[ixj] = a; }
            }
            __syncthreads();
        }
    }
    for (int p = tid; p < TT; p += 256) {
        unsigned idx = (unsigned)s[p];
        stbuf[(size_t)bh * TT + p]   = idx % TPAD;     // sticker % t
        undo [(size_t)bh * TT + idx] = (unsigned)p;    // argsort(sticker)
    }
}

// ----------------------------- chunked attention: one wave per (bh, chunk c)
__global__ __launch_bounds__(256)
void k_chunk_attn(const float* __restrict__ qkbuf, const float* __restrict__ vbuf,
                  const unsigned* __restrict__ stbuf,
                  float* __restrict__ so, float* __restrict__ slse) {
    __shared__ float kv[8][8][DH];
    __shared__ float vv[8][8][DH];
    __shared__ float pm[8][4][8];
    __shared__ float rn[8][8];
    int bh   = blockIdx.x;
    int w    = threadIdx.x >> 5, lane = threadIdx.x & 31;
    int c    = blockIdx.y * 8 + w;               // 0..519
    int cprv = (c + NCH - 1) % NCH;
    int b    = bh >> 3, head = bh & 7;
    const unsigned* st = stbuf + (size_t)bh * TT;

    // load 8 rows (chunk c rows 0-3, chunk c-1 rows 4-7) of qk and v
    for (int r = 0; r < 8; ++r) {
        int j = (r < 4) ? (c * BKT + r) : (cprv * BKT + (r - 4));
        unsigned t = st[j];
        size_t base = ((size_t)(b * TPAD + (int)t)) * D_ + head * DH;
        kv[w][r][lane]      = qkbuf[base + lane];
        kv[w][r][lane + 32] = qkbuf[base + lane + 32];
        vv[w][r][lane]      = vbuf[base + lane];
        vv[w][r][lane + 32] = vbuf[base + lane + 32];
    }
    __syncthreads();
    if (lane < 8) {
        float ss = 0.f;
#pragma unroll
        for (int f = 0; f < DH; ++f) ss += kv[w][lane][f] * kv[w][lane][f];
        rn[w][lane] = rsqrtf(fmaxf(ss, 1e-24f));
    }
    __syncthreads();

    int i  = lane >> 3;       // query row 0..3
    int jj = lane & 7;        // key row 0..7
    float d = 0.f;
#pragma unroll
    for (int f = 0; f < DH; ++f) d += kv[w][i][f] * kv[w][jj][f];
    d *= rn[w][jj] * 0.125f;  // dh^-0.5
    unsigned qt = st[c * BKT + i];
    int jrow    = (jj < 4) ? (c * BKT + jj) : (cprv * BKT + (jj - 4));
    unsigned kt = st[jrow];
    if (qt == kt) d = -5.0e4f;

    float mx = d;
#pragma unroll
    for (int o = 4; o; o >>= 1) mx = fmaxf(mx, __shfl_xor(mx, o, 8));
    float e  = expf(d - mx);
    float se = e;
#pragma unroll
    for (int o = 4; o; o >>= 1) se += __shfl_xor(se, o, 8);
    float lse = mx + logf(se);
    pm[w][i][jj] = e / se;     // exp(d - lse)
    if (jj == 0) slse[(size_t)bh * TT + c * BKT + i] = lse;
    __syncthreads();

    int g = lane & 7;          // output column group
    float acc[8] = {0.f,0.f,0.f,0.f,0.f,0.f,0.f,0.f};
#pragma unroll
    for (int j2 = 0; j2 < 8; ++j2) {
        float p = pm[w][i][j2];
#pragma unroll
        for (int u = 0; u < 8; ++u) acc[u] += p * vv[w][j2][g * 8 + u];
    }
    float* op = so + ((size_t)bh * TT + c * BKT + i) * DH + g * 8;
#pragma unroll
    for (int u = 0; u < 8; ++u) op[u] = acc[u];
}

// ----------------------------- combine hash rounds, write [B,520,D] head slab
__global__ __launch_bounds__(256)
void k_combine(const float* __restrict__ so, const float* __restrict__ slse,
               const unsigned* __restrict__ undo, float* __restrict__ attnh) {
    int bh = blockIdx.x;
    int tk = threadIdx.x >> 6;            // 0..3
    int f  = threadIdx.x & 63;
    int t  = blockIdx.y * 4 + tk;         // 0..519
    int b  = bh >> 3, head = bh & 7;
    unsigned idx[NHASH]; float lo[NHASH];
    float m = -3.0e38f;
#pragma unroll
    for (int r = 0; r < NHASH; ++r) {
        idx[r] = undo[(size_t)bh * TT + r * TPAD + t];
        lo[r]  = slse[(size_t)bh * TT + idx[r]];
        m = fmaxf(m, lo[r]);
    }
    float sw = 0.f, o = 0.f;
#pragma unroll
    for (int r = 0; r < NHASH; ++r) {
        float wr = expf(lo[r] - m);
        sw += wr;
        o  += wr * so[((size_t)bh * TT + idx[r]) * DH + f];
    }
    attnh[((size_t)(b * TPAD + t)) * D_ + head * DH + f] = o / sw;
}

// ------------------------------------------- residual + LayerNorm (rows < 516)
__global__ __launch_bounds__(256)
void k_ln_res(float* __restrict__ x, const float* __restrict__ y,
              const float* __restrict__ g, const float* __restrict__ bb) {
    __shared__ float s1[256], s2[256];
    int r = blockIdx.x;                 // 0..B*516-1
    int b = r / NTOK, t = r % NTOK;
    size_t base = ((size_t)(b * TPAD + t)) * D_;
    int tid = threadIdx.x;
    float v0 = x[base + tid]       + y[base + tid];
    float v1 = x[base + tid + 256] + y[base + tid + 256];
    s1[tid] = v0 + v1;
    s2[tid] = v0 * v0 + v1 * v1;
    __syncthreads();
    for (int s = 128; s > 0; s >>= 1) {
        if (tid < s) { s1[tid] += s1[tid + s]; s2[tid] += s2[tid + s]; }
        __syncthreads();
    }
    float mean = s1[0] * (1.f / D_);
    float var  = s2[0] * (1.f / D_) - mean * mean;
    float rs   = rsqrtf(var + 1e-5f);
    x[base + tid]       = (v0 - mean) * rs * g[tid]       + bb[tid];
    x[base + tid + 256] = (v1 - mean) * rs * g[tid + 256] + bb[tid + 256];
}

// ----------------------------------------- final LN (row 515) + linear head
__global__ __launch_bounds__(256)
void k_final(const float* __restrict__ x, const float* __restrict__ gF,
             const float* __restrict__ bF, const float* __restrict__ Wp,
             const float* __restrict__ bp, float* __restrict__ dec) {
    __shared__ float s1[256], s2[256];
    int b = blockIdx.x, tid = threadIdx.x;
    size_t base = ((size_t)(b * TPAD + (NTOK - 1))) * D_;
    float v0 = x[base + tid], v1 = x[base + tid + 256];
    s1[tid] = v0 + v1;
    s2[tid] = v0 * v0 + v1 * v1;
    __syncthreads();
    for (int s = 128; s > 0; s >>= 1) {
        if (tid < s) { s1[tid] += s1[tid + s]; s2[tid] += s2[tid + s]; }
        __syncthreads();
    }
    float mean = s1[0] * (1.f / D_);
    float var  = s2[0] * (1.f / D_) - mean * mean;
    float rs   = rsqrtf(var + 1e-5f);
    __syncthreads();
    float n0 = (v0 - mean) * rs * gF[tid]       + bF[tid];
    float n1 = (v1 - mean) * rs * gF[tid + 256] + bF[tid + 256];
    s1[tid] = n0 * Wp[tid] + n1 * Wp[tid + 256];
    __syncthreads();
    for (int s = 128; s > 0; s >>= 1) {
        if (tid < s) s1[tid] += s1[tid + s];
        __syncthreads();
    }
    if (tid == 0) dec[b] = s1[0] + bp[0];
}

// ------------------------------------------------- broadcast denorm [B,B,E]
__global__ __launch_bounds__(256)
void k_output(const float* __restrict__ dec, const float* __restrict__ stdb,
              const float* __restrict__ meanb, float* __restrict__ out) {
    int i  = blockIdx.x * 256 + threadIdx.x;     // B*B*E
    int e  = i % E_;
    int j  = (i / E_) % B_;
    int bi = i / (E_ * B_);
    out[i] = dec[j] * stdb[bi * E_ + e] + meanb[bi * E_ + e];
}

// ============================================================================
extern "C" void kernel_launch(void* const* d_in, const int* in_sizes, int n_in,
                              void* d_out, int out_size, void* d_ws, size_t ws_size,
                              hipStream_t stream) {
    (void)in_sizes; (void)n_in; (void)out_size; (void)ws_size;
    const float* x_enc  = (const float*)d_in[0];
    const float* x_mark = (const float*)d_in[1];
    const float* W_emb  = (const float*)d_in[2];
    const float* b_emb  = (const float*)d_in[3];
    const float* Wqk    = (const float*)d_in[4];
    const float* Wv     = (const float*)d_in[5];
    const float* Wo     = (const float*)d_in[6];
    const float* bo     = (const float*)d_in[7];
    const float* Wc1    = (const float*)d_in[8];
    const float* bc1    = (const float*)d_in[9];
    const float* Wc2    = (const float*)d_in[10];
    const float* bc2    = (const float*)d_in[11];
    const float* g1     = (const float*)d_in[12];
    const float* b1     = (const float*)d_in[13];
    const float* g2     = (const float*)d_in[14];
    const float* b2     = (const float*)d_in[15];
    const float* gF     = (const float*)d_in[16];
    const float* bF     = (const float*)d_in[17];
    const float* Wp     = (const float*)d_in[18];
    const float* bp     = (const float*)d_in[19];
    const float* rot    = (const float*)d_in[20];

    char* wsp = (char*)d_ws;
    auto alloc = [&](size_t bytes) -> void* {
        void* p = (void*)wsp;
        wsp += (bytes + 255) & ~(size_t)255;
        return p;
    };
    float* meanb = (float*)alloc((size_t)B_ * E_ * 4);
    float* stdb  = (float*)alloc((size_t)B_ * E_ * 4);
    float* x     = (float*)alloc((size_t)ROWS * D_ * 4);
    float* qk    = (float*)alloc((size_t)ROWS * D_ * 4);
    float* v     = (float*)alloc((size_t)ROWS * D_ * 4);   // also tok staging
    float* attnh = (float*)alloc((size_t)ROWS * D_ * 4);
    float* tmp   = (float*)alloc((size_t)ROWS * D_ * 4);
    float* big   = (float*)alloc((size_t)ROWS * DFF_ * 4); // so (attn) / ff1 (FFN)
    unsigned long long* sortb = (unsigned long long*)alloc((size_t)BH * SORTN * 8);
    unsigned* stb  = (unsigned*)alloc((size_t)BH * TT * 4);
    unsigned* und  = (unsigned*)alloc((size_t)BH * TT * 4);
    float*    slse = (float*)alloc((size_t)BH * TT * 4);
    float*    dec  = (float*)alloc((size_t)B_ * 4);

    // RevIN stats
    k_stats<<<dim3(B_, E_ / 256), 256, 0, stream>>>(x_enc, meanb, stdb);
    // tokens (aliased into v buffer) + embedding GEMM
    float* tok = v;
    k_build_tok<<<(ROWS * D_) / 256, 256, 0, stream>>>(x_enc, x_mark, meanb, stdb, tok);
    k_gemm<<<dim3(D_ / 64, ROWS / 128), 256, 0, stream>>>(tok, W_emb, b_emb, x,
                                                          ROWS, D_, D_, 0);
    k_zero_pad<<<(B_ * 4 * D_) / 256, 256, 0, stream>>>(x);

    for (int l = 0; l < NLAY; ++l) {
        const float* Wqk_l = Wqk + (size_t)l * D_ * D_;
        const float* Wv_l  = Wv  + (size_t)l * D_ * D_;
        const float* Wo_l  = Wo  + (size_t)l * D_ * D_;
        const float* bo_l  = bo  + (size_t)l * D_;
        const float* Wc1_l = Wc1 + (size_t)l * DFF_ * D_;
        const float* bc1_l = bc1 + (size_t)l * DFF_;
        const float* Wc2_l = Wc2 + (size_t)l * D_ * DFF_;
        const float* bc2_l = bc2 + (size_t)l * D_;
        const float* rot_l = rot + (size_t)l * DH * NHASH * (NB / 2);

        // shared-QK and V projections (over all 520 rows, pads are zero)
        k_gemm<<<dim3(D_ / 64, ROWS / 128), 256, 0, stream>>>(x, Wqk_l, nullptr, qk,
                                                              ROWS, D_, D_, 0);
        k_gemm<<<dim3(D_ / 64, ROWS / 128), 256, 0, stream>>>(x, Wv_l, nullptr, v,
                                                              ROWS, D_, D_, 0);
        // LSH attention
        k_buckets<<<dim3(BH, SORTN / 256), 256, 0, stream>>>(qk, rot_l, sortb);
        k_sort<<<BH, 256, 0, stream>>>(sortb, stb, und);
        k_chunk_attn<<<dim3(BH, NCH / 8), 256, 0, stream>>>(qk, v, stb, big, slse);
        k_combine<<<dim3(BH, NCH / 4), 256, 0, stream>>>(big, slse, und, attnh);
        // output projection + residual LN
        k_gemm<<<dim3(D_ / 64, ROWS / 128), 256, 0, stream>>>(attnh, Wo_l, bo_l, tmp,
                                                              ROWS, D_, D_, 0);
        k_ln_res<<<B_ * NTOK, 256, 0, stream>>>(x, tmp, g1 + (size_t)l * D_,
                                                b1 + (size_t)l * D_);
        // FFN (big buffer reused for ff1) + residual LN
        k_gemm<<<dim3(DFF_ / 64, ROWS / 128), 256, 0, stream>>>(x, Wc1_l, bc1_l, big,
                                                                ROWS, DFF_, D_, 1);
        k_gemm<<<dim3(D_ / 64, ROWS / 128), 256, 0, stream>>>(big, Wc2_l, bc2_l, tmp,
                                                              ROWS, D_, DFF_, 0);
        k_ln_res<<<B_ * NTOK, 256, 0, stream>>>(x, tmp, g2 + (size_t)l * D_,
                                                b2 + (size_t)l * D_);
    }

    k_final<<<B_, 256, 0, stream>>>(x, gF, bF, Wp, bp, dec);
    k_output<<<(B_ * B_ * E_) / 256, 256, 0, stream>>>(dec, stdb, meanb, (float*)d_out);
}
